// GAT_24661702214222
// MI455X (gfx1250) — compile-verified, matrix operand order checked by reading, not run on hardware
//
#include <hip/hip_runtime.h>
#include <hip/hip_bf16.h>
#include <math.h>

// ---------------------------------------------------------------------------
// GAT (2-layer, PyG-style) for MI455X / gfx1250, wave32 + WMMA.
// ---------------------------------------------------------------------------

typedef __attribute__((ext_vector_type(16))) _Float16 v16h;
typedef __attribute__((ext_vector_type(8)))  _Float16 v8h;
typedef __attribute__((ext_vector_type(8)))  float    v8f;

#define NEG_SLOPE 0.2f
#define EPS_DEN   1e-16f

// float atomic max via monotone int/uint reinterpretation (native L2 atomics)
__device__ __forceinline__ void atomic_max_f32(float* addr, float val) {
  if (!(val < 0.0f)) {
    atomicMax(reinterpret_cast<int*>(addr), __float_as_int(val));
  } else {
    atomicMin(reinterpret_cast<unsigned int*>(addr), __float_as_uint(val));
  }
}

// ---------------------------- small utility kernels ------------------------
__global__ void k_cast_f32_to_f16(const float* __restrict__ in,
                                  _Float16* __restrict__ out, long long n) {
  long long i = (long long)blockIdx.x * blockDim.x + threadIdx.x;
  if (i < n) out[i] = (_Float16)in[i];
}

// W [K,Nc] f32 row-major -> WT [Nc,K] f16 row-major (so B-fragment loads are contiguous)
__global__ void k_transpose_cast(const float* __restrict__ W,
                                 _Float16* __restrict__ WT, int K, int Nc) {
  int i = blockIdx.x * blockDim.x + threadIdx.x;
  if (i >= K * Nc) return;
  int k = i / Nc, n = i - k * Nc;
  WT[(size_t)n * K + k] = (_Float16)W[i];
}

__global__ void k_fill(float* __restrict__ p, float v, long long n) {
  long long i = (long long)blockIdx.x * blockDim.x + threadIdx.x;
  if (i < n) p[i] = v;
}

// ---------------------------- WMMA GEMM ------------------------------------
// ISA 16-bit A layout (16x32): lanes 0-15 hold row M=lane, K=0..7 & 16..23;
// lanes 16-31 hold row M=lane-16, K=8..15 & 24..31.  Row-major fp16 source
// means each lane needs exactly two contiguous 16B loads per K-step of 32.
__device__ __forceinline__ v16h load_frag16(const _Float16* __restrict__ base,
                                            int ld, int r0, int k0, int lane) {
  const int row = r0 + (lane & 15);
  const int kb  = k0 + ((lane >> 4) << 3);          // +0 or +8
  const _Float16* p = base + (size_t)row * ld + kb;
  v8h lo = *(const v8h*)(p);                        // K = kb .. kb+7
  v8h hi = *(const v8h*)(p + 16);                   // K = kb+16 .. kb+23
  v16h r;
#pragma unroll
  for (int i = 0; i < 8; ++i) { r[i] = lo[i]; r[i + 8] = hi[i]; }
  return r;
}

// C = A[M,K] * BT[Nc,K]^T, f32 accum.  One 16x16 tile per wave32.
__global__ void k_wmma_gemm(const _Float16* __restrict__ A,
                            const _Float16* __restrict__ BT,
                            float* __restrict__ C,
                            int M, int K, int Nc) {
  const int lane = threadIdx.x & 31;
  const int wavesPerBlock = blockDim.x >> 5;
  const int g = blockIdx.x * wavesPerBlock + (threadIdx.x >> 5);
  const int tiles_n = Nc >> 4;
  const int tiles_m = M >> 4;
  if (g >= tiles_m * tiles_n) return;               // wave-uniform guard: EXEC all-1 at WMMA
  const int m0 = (g / tiles_n) << 4;
  const int n0 = (g % tiles_n) << 4;

  v8f acc = {};
  for (int k0 = 0; k0 < K; k0 += 32) {
    v16h a = load_frag16(A,  K, m0, k0, lane);
    v16h b = load_frag16(BT, K, n0, k0, lane);
    if (k0 + 32 < K) {                              // global_prefetch_b8 next K-slice
      __builtin_prefetch(A  + (size_t)(m0 + (lane & 15)) * K + (k0 + 32), 0, 1);
      __builtin_prefetch(BT + (size_t)(n0 + (lane & 15)) * K + (k0 + 32), 0, 1);
    }
    acc = __builtin_amdgcn_wmma_f32_16x16x32_f16(
        /*neg_a=*/false, a, /*neg_b=*/false, b,
        /*c_mod=*/(short)0, acc, /*reuse_a=*/false, /*reuse_b=*/false);
  }
  // C/D layout: VGPR j -> row m0 + j (+8 for upper lane half), col = n0 + (lane&15)
  const int row_base = m0 + ((lane >> 4) << 3);
  const int col = n0 + (lane & 15);
#pragma unroll
  for (int j = 0; j < 8; ++j)
    C[(size_t)(row_base + j) * Nc + col] = acc[j];
}

// ---------------------------- attention pieces -----------------------------
// a_src[n,h] = sum_c h[n,h,c]*att_src[h,c]  (likewise a_dst)
__global__ void k_attn_coeff(const float* __restrict__ h,
                             const float* __restrict__ att_src,
                             const float* __restrict__ att_dst,
                             float* __restrict__ asrc, float* __restrict__ adst,
                             int Nn, int H, int Cc) {
  int idx = blockIdx.x * blockDim.x + threadIdx.x;
  if (idx >= Nn * H) return;
  int n = idx / H, hh = idx - n * H;
  const float* hp = h + (size_t)n * H * Cc + (size_t)hh * Cc;
  const float* as = att_src + hh * Cc;
  const float* ad = att_dst + hh * Cc;
  float s = 0.f, d = 0.f;
  for (int c = 0; c < Cc; ++c) { float v = hp[c]; s += v * as[c]; d += v * ad[c]; }
  asrc[idx] = s;
  adst[idx] = d;
}

__device__ __forceinline__ void edge_nodes(int e, const int* __restrict__ src,
                                           const int* __restrict__ dst,
                                           int E0, int& s, int& d) {
  if (e < E0) { s = src[e]; d = dst[e]; }
  else        { s = e - E0; d = s; }                // appended self-loops
}

__device__ __forceinline__ float leaky(float v) {
  return v > 0.f ? v : NEG_SLOPE * v;
}

// pass 1: segment max of leaky_relu(a_src[s]+a_dst[d]) over dst
__global__ void k_edge_max(const float* __restrict__ asrc, const float* __restrict__ adst,
                           const int* __restrict__ src, const int* __restrict__ dst,
                           float* __restrict__ m, int E0, int Nn, int H) {
  long long idx = (long long)blockIdx.x * blockDim.x + threadIdx.x;
  long long total = (long long)(E0 + Nn) * H;
  if (idx >= total) return;
  int e = (int)(idx / H), hh = (int)(idx % H);
  int s, d; edge_nodes(e, src, dst, E0, s, d);
  float ev = leaky(asrc[s * H + hh] + adst[d * H + hh]);
  atomic_max_f32(&m[d * H + hh], ev);
}

// pass 2: segment sum of exp(e - m[dst])
__global__ void k_edge_sum(const float* __restrict__ asrc, const float* __restrict__ adst,
                           const int* __restrict__ src, const int* __restrict__ dst,
                           const float* __restrict__ m, float* __restrict__ sbuf,
                           int E0, int Nn, int H) {
  long long idx = (long long)blockIdx.x * blockDim.x + threadIdx.x;
  long long total = (long long)(E0 + Nn) * H;
  if (idx >= total) return;
  int e = (int)(idx / H), hh = (int)(idx % H);
  int s, d; edge_nodes(e, src, dst, E0, s, d);
  float ev = leaky(asrc[s * H + hh] + adst[d * H + hh]);
  atomicAdd(&sbuf[d * H + hh], __expf(ev - m[d * H + hh]));
}

// pass 3: out[d,h,c] += h[s,h,c] * alpha  (alpha recomputed; h & out live in L2)
__global__ void k_edge_agg(const float* __restrict__ h,
                           const float* __restrict__ asrc, const float* __restrict__ adst,
                           const int* __restrict__ src, const int* __restrict__ dst,
                           const float* __restrict__ m, const float* __restrict__ sbuf,
                           float* __restrict__ out, int E0, int Nn, int H, int Cc) {
  long long idx = (long long)blockIdx.x * blockDim.x + threadIdx.x;
  const int F = H * Cc;
  long long total = (long long)(E0 + Nn) * F;
  if (idx >= total) return;
  int f = (int)(idx % F);
  int e = (int)(idx / F);
  int hh = f / Cc;
  int s, d; edge_nodes(e, src, dst, E0, s, d);
  float ev = leaky(asrc[s * H + hh] + adst[d * H + hh]);
  float alpha = __expf(ev - m[d * H + hh]) / (sbuf[d * H + hh] + EPS_DEN);
  atomicAdd(&out[(size_t)d * F + f], h[(size_t)s * F + f] * alpha);
}

// h2 = fp16(elu(out1 + b1))  — feeds GEMM2
__global__ void k_elu_cast(const float* __restrict__ in, const float* __restrict__ bias,
                           _Float16* __restrict__ outh, long long n, int F) {
  long long i = (long long)blockIdx.x * blockDim.x + threadIdx.x;
  if (i >= n) return;
  float v = in[i] + bias[(int)(i % F)];
  v = v > 0.f ? v : (__expf(v) - 1.0f);
  outh[i] = (_Float16)v;
}

// y = log_softmax(out2 + b2) over Cc=16
__global__ void k_bias_logsoftmax(const float* __restrict__ in, const float* __restrict__ bias,
                                  float* __restrict__ out, int Nn, int Cc) {
  int n = blockIdx.x * blockDim.x + threadIdx.x;
  if (n >= Nn) return;
  float v[16];
  float mx = -3.4e38f;
  for (int c = 0; c < Cc; ++c) { v[c] = in[(size_t)n * Cc + c] + bias[c]; mx = fmaxf(mx, v[c]); }
  float sum = 0.f;
  for (int c = 0; c < Cc; ++c) sum += __expf(v[c] - mx);
  float lse = mx + __logf(sum);
  for (int c = 0; c < Cc; ++c) out[(size_t)n * Cc + c] = v[c] - lse;
}

// ---------------------------------------------------------------------------
extern "C" void kernel_launch(void* const* d_in, const int* in_sizes, int n_in,
                              void* d_out, int out_size, void* d_ws, size_t ws_size,
                              hipStream_t stream) {
  (void)n_in; (void)out_size; (void)ws_size;
  const float* x   = (const float*)d_in[0];
  const int*   ei  = (const int*)d_in[1];
  const float* W1  = (const float*)d_in[2];
  const float* b1  = (const float*)d_in[3];
  const float* as1 = (const float*)d_in[4];
  const float* ad1 = (const float*)d_in[5];
  const float* W2  = (const float*)d_in[6];
  const float* b2  = (const float*)d_in[7];
  const float* as2 = (const float*)d_in[8];
  const float* ad2 = (const float*)d_in[9];
  float* y = (float*)d_out;

  const int Nn = 50000;
  const int Fin = 128;
  const int H1 = 8, C1 = 32, F1 = H1 * C1;    // 256
  const int H2 = 1, C2 = 16, F2 = H2 * C2;    // 16
  const int E0 = in_sizes[1] / 2;             // 800000
  const int* srcI = ei;
  const int* dstI = ei + E0;

  // workspace bump allocator (256B aligned so 16B vector loads are aligned)
  char* p = (char*)d_ws;
  auto carve = [&](size_t bytes) -> char* {
    char* r = p; p += (bytes + 255) & ~(size_t)255; return r;
  };
  _Float16* xh   = (_Float16*)carve((size_t)Nn * Fin * 2);
  _Float16* w1t  = (_Float16*)carve((size_t)F1 * Fin * 2);
  float*    h1   = (float*)carve((size_t)Nn * F1 * 4);
  float*    a_s1 = (float*)carve((size_t)Nn * H1 * 4);
  float*    a_d1 = (float*)carve((size_t)Nn * H1 * 4);
  float*    m1   = (float*)carve((size_t)Nn * H1 * 4);
  float*    s1   = (float*)carve((size_t)Nn * H1 * 4);
  float*    o1   = (float*)carve((size_t)Nn * F1 * 4);
  _Float16* h2h  = (_Float16*)carve((size_t)Nn * F1 * 2);
  _Float16* w2t  = (_Float16*)carve((size_t)F2 * F1 * 2);
  float*    h2   = (float*)carve((size_t)Nn * F2 * 4);
  float*    a_s2 = (float*)carve((size_t)Nn * H2 * 4);
  float*    a_d2 = (float*)carve((size_t)Nn * H2 * 4);
  float*    m2   = (float*)carve((size_t)Nn * H2 * 4);
  float*    s2   = (float*)carve((size_t)Nn * H2 * 4);
  float*    o2   = (float*)carve((size_t)Nn * F2 * 4);

  const int B = 256;  // 8 wave32 per block
  auto blocks = [](long long n, int b) { return (unsigned)((n + b - 1) / b); };

  // ---------------- layer 1 ----------------
  k_cast_f32_to_f16<<<blocks((long long)Nn * Fin, B), B, 0, stream>>>(x, xh, (long long)Nn * Fin);
  k_transpose_cast<<<blocks((long long)Fin * F1, B), B, 0, stream>>>(W1, w1t, Fin, F1);
  {
    long long waves = (long long)(Nn / 16) * (F1 / 16);      // one tile per wave
    k_wmma_gemm<<<blocks(waves * 32, B), B, 0, stream>>>(xh, w1t, h1, Nn, Fin, F1);
  }
  k_attn_coeff<<<blocks((long long)Nn * H1, B), B, 0, stream>>>(h1, as1, ad1, a_s1, a_d1, Nn, H1, C1);
  k_fill<<<blocks((long long)Nn * H1, B), B, 0, stream>>>(m1, -INFINITY, (long long)Nn * H1);
  k_fill<<<blocks((long long)Nn * H1, B), B, 0, stream>>>(s1, 0.f, (long long)Nn * H1);
  k_fill<<<blocks((long long)Nn * F1, B), B, 0, stream>>>(o1, 0.f, (long long)Nn * F1);
  {
    long long eh = (long long)(E0 + Nn) * H1;
    k_edge_max<<<blocks(eh, B), B, 0, stream>>>(a_s1, a_d1, srcI, dstI, m1, E0, Nn, H1);
    k_edge_sum<<<blocks(eh, B), B, 0, stream>>>(a_s1, a_d1, srcI, dstI, m1, s1, E0, Nn, H1);
    long long ehc = (long long)(E0 + Nn) * F1;
    k_edge_agg<<<blocks(ehc, B), B, 0, stream>>>(h1, a_s1, a_d1, srcI, dstI, m1, s1, o1, E0, Nn, H1, C1);
  }
  k_elu_cast<<<blocks((long long)Nn * F1, B), B, 0, stream>>>(o1, b1, h2h, (long long)Nn * F1, F1);

  // ---------------- layer 2 ----------------
  k_transpose_cast<<<blocks((long long)F1 * F2, B), B, 0, stream>>>(W2, w2t, F1, F2);
  {
    long long waves = (long long)(Nn / 16) * (F2 / 16);
    k_wmma_gemm<<<blocks(waves * 32, B), B, 0, stream>>>(h2h, w2t, h2, Nn, F1, F2);
  }
  k_attn_coeff<<<blocks((long long)Nn * H2, B), B, 0, stream>>>(h2, as2, ad2, a_s2, a_d2, Nn, H2, C2);
  k_fill<<<blocks((long long)Nn * H2, B), B, 0, stream>>>(m2, -INFINITY, (long long)Nn * H2);
  k_fill<<<blocks((long long)Nn * H2, B), B, 0, stream>>>(s2, 0.f, (long long)Nn * H2);
  k_fill<<<blocks((long long)Nn * F2, B), B, 0, stream>>>(o2, 0.f, (long long)Nn * F2);
  {
    long long eh = (long long)(E0 + Nn) * H2;
    k_edge_max<<<blocks(eh, B), B, 0, stream>>>(a_s2, a_d2, srcI, dstI, m2, E0, Nn, H2);
    k_edge_sum<<<blocks(eh, B), B, 0, stream>>>(a_s2, a_d2, srcI, dstI, m2, s2, E0, Nn, H2);
    long long ehc = (long long)(E0 + Nn) * F2;
    k_edge_agg<<<blocks(ehc, B), B, 0, stream>>>(h2, a_s2, a_d2, srcI, dstI, m2, s2, o2, E0, Nn, H2, C2);
  }
  k_bias_logsoftmax<<<blocks(Nn, B), B, 0, stream>>>(o2, b2, y, Nn, C2);
}
